// MultiheadLatentAttention_14456859919084
// MI455X (gfx1250) — matrix operand order repaired and checked by reading, not compiled, not run on hardware
//
#include <hip/hip_runtime.h>

// ---------------- problem constants (match reference) ----------------
#define C_B   2
#define C_S   2048
#define C_L   2048
#define C_DQ  1024
#define C_H   16
#define C_K   64
#define C_DC  128
#define C_DCQ 32
#define C_R   64
#define C_HK  (C_H * C_K)   // 1024

typedef __attribute__((ext_vector_type(16))) _Float16 v16h;
typedef __attribute__((ext_vector_type(8)))  float    v8f;

// ============ WMMA fragment loaders (per CDNA5 ISA) ============
// A fragment: 16x32 f16, row-major source, tile at (row0, k0).
// Per-lane: two contiguous 16B chunks -> 2x global_load_b128.
__device__ __forceinline__ v16h load_a_frag(const _Float16* __restrict__ A,
                                            int lda, int row0, int k0) {
  int lane = threadIdx.x & 31;
  int m = lane & 15;
  int kbase = (lane < 16) ? 0 : 16;
  const _Float16* p = A + (size_t)(row0 + m) * lda + k0;
  v16h f;
#pragma unroll
  for (int v = 0; v < 8; ++v) {
    int k = kbase + 2 * v;
    f[2 * v]     = p[k];
    f[2 * v + 1] = p[k + 1];
  }
  return f;
}

// B fragment from TRANSPOSED storage BT (N x K row-major):
// logical B(k, n) = BT[(col0+n)*ldbt + k0 + k].
// Per-lane: 16 contiguous f16 (32B) -> 2x global_load_b128.
__device__ __forceinline__ v16h load_bt_frag(const _Float16* __restrict__ BT,
                                             int ldbt, int k0, int col0) {
  int lane = threadIdx.x & 31;
  int n = lane & 15;
  int kbase = (lane < 16) ? 0 : 16;
  const _Float16* p = BT + (size_t)(col0 + n) * ldbt + k0 + kbase;
  v16h f;
#pragma unroll
  for (int v = 0; v < 16; ++v) f[v] = p[v];
  return f;
}

// B fragment from an LDS-resident BT panel ([n][k] layout, leading dim ld).
// Per-lane contiguous 32B -> 2x ds_load_b128.
__device__ __forceinline__ v16h lds_panel_frag(const _Float16* panel, int ld,
                                               int k0, int col0) {
  int lane = threadIdx.x & 31;
  int n = lane & 15;
  int kbase = (lane < 16) ? 0 : 16;
  const _Float16* p = panel + (col0 + n) * ld + k0 + kbase;
  v16h f;
#pragma unroll
  for (int v = 0; v < 16; ++v) f[v] = p[v];
  return f;
}

// ---- CDNA5 async global->LDS (ASYNCcnt-tracked, CDNA5 ISA)
__device__ __forceinline__ void async_b128(unsigned lds_off,
                                           const _Float16* gaddr) {
  asm volatile("global_load_async_to_lds_b128 %0, %1, off"
               :: "v"(lds_off), "v"(gaddr) : "memory");
}

// ======== tiled GEMM for the big projections: C = A x BT^T ========
// 256 threads (8 waves); block tile 128x64; wave tile 16x64 (4 accumulators).
// BT panel (64 cols x 32 k = 4KB) double-buffered in LDS via async loads.
template <typename OutT>
__global__ void __launch_bounds__(256)
wmma_gemm_tiled_kernel(const _Float16* __restrict__ A,
                       const _Float16* __restrict__ BT, OutT* __restrict__ C,
                       int M, int N, int Kd, int lda, int ldbt, int ldc) {
  __shared__ __align__(16) _Float16 smem[2][64 * 32];
  int wv = threadIdx.x >> 5;
  int lane = threadIdx.x & 31;
  int row0 = blockIdx.x * 128 + wv * 16;
  int col0 = blockIdx.y * 64;

  auto stage = [&](int k0, int buf) {
    int t = threadIdx.x;          // 256 threads, one b128 each
    int n = t >> 2;               // 0..63
    int ch = (t & 3) * 8;         // 0,8,16,24 (f16 elems)
    const _Float16* g = BT + (size_t)(col0 + n) * ldbt + k0 + ch;
    async_b128((unsigned)(size_t)(&smem[buf][n * 32 + ch]), g);
  };

  v8f acc[4];
  v8f vzero = {};
#pragma unroll
  for (int i = 0; i < 4; ++i) acc[i] = vzero;

  stage(0, 0);
  int buf = 0;
  for (int k0 = 0; k0 < Kd; k0 += 32) {
    bool more = (k0 + 32 < Kd);
    if (more) stage(k0 + 32, buf ^ 1);
    if (more) asm volatile("s_wait_asynccnt 0x1" ::: "memory");
    else      asm volatile("s_wait_asynccnt 0x0" ::: "memory");
    __syncthreads();
    v16h a = load_a_frag(A, lda, row0, k0);
#pragma unroll
    for (int ct = 0; ct < 4; ++ct) {
      v16h b = lds_panel_frag(&smem[buf][0], 32, 0, ct * 16);
      acc[ct] = __builtin_amdgcn_wmma_f32_16x16x32_f16(false, a, false, b,
                                                       (short)0, acc[ct],
                                                       false, false);
    }
    __syncthreads();
    buf ^= 1;
  }

  int n = lane & 15;
  int mbase = (lane < 16) ? 0 : 8;
#pragma unroll
  for (int ct = 0; ct < 4; ++ct)
#pragma unroll
    for (int v = 0; v < 8; ++v)
      C[(size_t)(row0 + mbase + v) * ldc + col0 + ct * 16 + n] =
          (OutT)acc[ct][v];
}

// ============ generic batched WMMA GEMM: C = A(f16) x BT(f16)^T ============
template <typename OutT>
__global__ void __launch_bounds__(128)
wmma_gemm_kernel(const _Float16* __restrict__ A, const _Float16* __restrict__ BT,
                 OutT* __restrict__ C, int M, int N, int Kd, int lda, int ldbt,
                 int ldc, long sA0, long sA1, long sB0, long sB1, long sC0,
                 long sC1, int n1) {
  int wv = threadIdx.x >> 5;
  int tilesN = N >> 4;
  int tilesM = M >> 4;
  int t = blockIdx.x * 4 + wv;
  if (t >= tilesM * tilesN) return;
  int b0 = blockIdx.y / n1, b1 = blockIdx.y % n1;
  const _Float16* Ab = A + (size_t)b0 * sA0 + (size_t)b1 * sA1;
  const _Float16* Bb = BT + (size_t)b0 * sB0 + (size_t)b1 * sB1;
  OutT* Cb = C + (size_t)b0 * sC0 + (size_t)b1 * sC1;
  int tm = t / tilesN, tn = t % tilesN;
  int row0 = tm << 4, col0 = tn << 4;

  v8f acc = {};
  for (int k0 = 0; k0 < Kd; k0 += 32) {
    if (k0 + 32 < Kd)  // L2 prefetch of next A K-slab (global_prefetch_b8)
      __builtin_prefetch(Ab + (size_t)(row0 + (threadIdx.x & 15)) * lda + k0 + 32, 0, 1);
    v16h a = load_a_frag(Ab, lda, row0, k0);
    v16h b = load_bt_frag(Bb, ldbt, k0, col0);
    acc = __builtin_amdgcn_wmma_f32_16x16x32_f16(false, a, false, b, (short)0,
                                                 acc, false, false);
  }
  int lane = threadIdx.x & 31;
  int n = lane & 15;
  int mbase = (lane < 16) ? 0 : 8;
#pragma unroll
  for (int v = 0; v < 8; ++v)
    Cb[(size_t)(row0 + mbase + v) * ldc + col0 + n] = (OutT)acc[v];
}

// ============ fp32 -> f16 conversion ============
__global__ void cvt_kernel(const float* __restrict__ in,
                           _Float16* __restrict__ out, long n) {
  long t = (long)blockIdx.x * blockDim.x + threadIdx.x;
  if (t < n) out[t] = (_Float16)in[t];
}

// fp32 -> f16 with per-batch transpose: out[b][c*rows + r] = in[b][r*cols + c]
__global__ void cvt_t_kernel(const float* __restrict__ in,
                             _Float16* __restrict__ out, int rows, int cols,
                             long n) {
  long t = (long)blockIdx.x * blockDim.x + threadIdx.x;
  if (t >= n) return;
  int c = (int)(t % cols);
  long rb = t / cols;
  int r = (int)(rb % rows);
  long b = rb / rows;
  out[b * (long)rows * cols + (long)c * rows + r] = (_Float16)in[t];
}

// ============ RoPE: (BH, len, 64) f32 -> f16 ============
__global__ void rope_kernel(const float* __restrict__ in,
                            _Float16* __restrict__ out, int len, long total) {
  long t = (long)blockIdx.x * blockDim.x + threadIdx.x;
  if (t >= total) return;
  int i = (int)(t & 31);
  long pj = t >> 5;               // bh*len + p
  int p = (int)(pj % len);
  long base = pj * 64;
  float freq = __expf(-((float)(2 * i) / 64.0f) * __logf(10000.0f));
  float ang = (float)p * freq;
  float c = __cosf(ang), s = __sinf(ang);
  float x1 = in[base + i], x2 = in[base + 32 + i];
  out[base + i]      = (_Float16)(x1 * c - x2 * s);
  out[base + 32 + i] = (_Float16)(x1 * s + x2 * c);
}

// ============ fused attention: scores -> online softmax -> attn x V ============
// One block (8 waves) = 128 query rows of one (b,h); each wave owns 16 rows.
// Shared K/V panels (k_r 32x64, kv_c^T 128x32) double-buffered in LDS via
// async-to-LDS loads, consumed by all 8 waves (8x traffic reduction).
__global__ void __launch_bounds__(256)
flash_attn_kernel(const _Float16* __restrict__ qr, const _Float16* __restrict__ kr,
                  const _Float16* __restrict__ kvcT, _Float16* __restrict__ ctx) {
  __shared__ __align__(16) _Float16 skr[2][32 * 64];    // k_r panel  [l][r]
  __shared__ __align__(16) _Float16 skv[2][128 * 32];   // kv_c^T panel [c][l]
  __shared__ __align__(16) _Float16 sexp[8][16 * 32];   // per-wave exp staging

  int lane = threadIdx.x & 31;
  int wv = threadIdx.x >> 5;
  int blocksPerBH = C_S / 128;
  int bh = blockIdx.x / blocksPerBH;
  int t128 = blockIdx.x % blocksPerBH;
  int b = bh / C_H;
  int ts = t128 * 8 + wv;     // this wave's 16-row tile index within (b,h)

  const _Float16* qtile = qr + (size_t)bh * C_S * C_R + (size_t)ts * 16 * C_R;
  const _Float16* kbase = kr + (size_t)bh * C_L * C_R;
  const _Float16* vbT = kvcT + (size_t)b * C_DC * C_L;
  _Float16* ctile = ctx + (size_t)bh * C_S * C_DC + (size_t)ts * 16 * C_DC;

  // cooperative async stage of both panels for l-range [l0, l0+32)
  auto stage = [&](int l0, int buf) {
    int t = threadIdx.x;
    {
      int row = t >> 3;           // 0..31 (l)
      int ch = (t & 7) * 8;       // 0..56 (r)
      async_b128((unsigned)(size_t)(&skr[buf][row * 64 + ch]),
                 kbase + (size_t)(l0 + row) * C_R + ch);
    }
#pragma unroll
    for (int i = 0; i < 2; ++i) {
      int idx = t + i * 256;      // 0..511
      int c = idx >> 2;           // 0..127
      int ch = (idx & 3) * 8;     // 0..24 (l)
      async_b128((unsigned)(size_t)(&skv[buf][c * 32 + ch]),
                 vbT + (size_t)c * C_L + l0 + ch);
    }
  };

  v16h qa0 = load_a_frag(qtile, C_R, 0, 0);
  v16h qa1 = load_a_frag(qtile, C_R, 0, 32);

  float mrow[8], srow[8];
  v8f acc[8];
  v8f vzero = {};
#pragma unroll
  for (int i = 0; i < 8; ++i) { mrow[i] = -3.0e38f; srow[i] = 0.0f; acc[i] = vzero; }

  int n = lane & 15;
  int mbase = (lane < 16) ? 0 : 8;

  stage(0, 0);
  int buf = 0;
  for (int l0 = 0; l0 < C_L; l0 += 32) {
    bool more = (l0 + 32 < C_L);
    if (more) stage(l0 + 32, buf ^ 1);
    if (more) asm volatile("s_wait_asynccnt 0x3" ::: "memory");  // 3 pending/stage
    else      asm volatile("s_wait_asynccnt 0x0" ::: "memory");
    __syncthreads();

    // ---- scores: 16x32 tile via 4 WMMAs from LDS k_r panel ----
    v8f sc0 = {}, sc1 = {};
    v16h kb;
    kb = lds_panel_frag(&skr[buf][0], 64, 0, 0);
    sc0 = __builtin_amdgcn_wmma_f32_16x16x32_f16(false, qa0, false, kb, (short)0, sc0, false, false);
    kb = lds_panel_frag(&skr[buf][0], 64, 32, 0);
    sc0 = __builtin_amdgcn_wmma_f32_16x16x32_f16(false, qa1, false, kb, (short)0, sc0, false, false);
    kb = lds_panel_frag(&skr[buf][0], 64, 0, 16);
    sc1 = __builtin_amdgcn_wmma_f32_16x16x32_f16(false, qa0, false, kb, (short)0, sc1, false, false);
    kb = lds_panel_frag(&skr[buf][0], 64, 32, 16);
    sc1 = __builtin_amdgcn_wmma_f32_16x16x32_f16(false, qa1, false, kb, (short)0, sc1, false, false);

    // ---- online softmax per row (row = mbase+v, spread over 16 lanes) ----
#pragma unroll
    for (int v = 0; v < 8; ++v) {
      float x = fmaxf(sc0[v], sc1[v]);
#pragma unroll
      for (int msk = 8; msk; msk >>= 1) x = fmaxf(x, __shfl_xor(x, msk, 16));
      float mnew = fmaxf(mrow[v], x);
      float scale = __expf(mrow[v] - mnew);
      mrow[v] = mnew;
      float e0 = __expf(sc0[v] - mnew);
      float e1 = __expf(sc1[v] - mnew);
      float ssum = e0 + e1;
#pragma unroll
      for (int msk = 8; msk; msk >>= 1) ssum += __shfl_xor(ssum, msk, 16);
      srow[v] = srow[v] * scale + ssum;
#pragma unroll
      for (int ct = 0; ct < 8; ++ct) acc[ct][v] *= scale;
      int row = mbase + v;
      sexp[wv][row * 32 + n]      = (_Float16)e0;
      sexp[wv][row * 32 + 16 + n] = (_Float16)e1;
    }

    // ---- relayout exp tile (C-frag -> A-frag) through per-wave LDS ----
    v16h ea;
    {
      int m = lane & 15;
      int kb2 = (lane < 16) ? 0 : 16;
#pragma unroll
      for (int v = 0; v < 8; ++v) {
        int k = kb2 + 2 * v;
        ea[2 * v]     = sexp[wv][m * 32 + k];
        ea[2 * v + 1] = sexp[wv][m * 32 + k + 1];
      }
    }

    // ---- ctx accumulation: exp(16x32) x kv_c(32x128) from LDS V panel ----
#pragma unroll
    for (int ct = 0; ct < 8; ++ct) {
      v16h vb = lds_panel_frag(&skv[buf][0], 32, 0, ct * 16);
      acc[ct] = __builtin_amdgcn_wmma_f32_16x16x32_f16(false, ea, false, vb, (short)0, acc[ct], false, false);
    }
    __syncthreads();
    buf ^= 1;
  }

  // ---- normalize and store f16 ctx tile ----
#pragma unroll
  for (int v = 0; v < 8; ++v) {
    float inv = 1.0f / srow[v];
#pragma unroll
    for (int ct = 0; ct < 8; ++ct)
      ctile[(size_t)(mbase + v) * C_DC + ct * 16 + n] = (_Float16)(acc[ct][v] * inv);
  }
}

// ============================ host launch ============================
extern "C" void kernel_launch(void* const* d_in, const int* in_sizes, int n_in,
                              void* d_out, int out_size, void* d_ws,
                              size_t ws_size, hipStream_t stream) {
  (void)in_sizes; (void)n_in; (void)out_size; (void)ws_size;
  const float* hidden_q = (const float*)d_in[0];
  const float* kv_c     = (const float*)d_in[1];
  const float* W_q      = (const float*)d_in[2];
  const float* w_kc_q   = (const float*)d_in[3];
  const float* w_kc_kv  = (const float*)d_in[4];
  const float* W_qr     = (const float*)d_in[5];
  const float* W_kr     = (const float*)d_in[6];
  const float* W_out    = (const float*)d_in[7];
  float* out = (float*)d_out;

  char* ws = (char*)d_ws;
  size_t off = 0;
  auto take = [&](size_t elems, size_t esz) -> void* {
    void* p = ws + off;
    off += (elems * esz + 255) & ~(size_t)255;
    return p;
  };
  _Float16* hq_h     = (_Float16*)take((size_t)C_B * C_S * C_DQ, 2);
  _Float16* Wq_h     = (_Float16*)take((size_t)C_HK * C_DQ, 2);      // BT for q-proj
  _Float16* q_h      = (_Float16*)take((size_t)C_B * C_S * C_HK, 2);
  _Float16* wkcq_h   = (_Float16*)take((size_t)C_H * C_K * C_DCQ, 2); // BT for WcombT
  _Float16* WqrT_h   = (_Float16*)take((size_t)C_H * C_R * C_DCQ, 2);
  _Float16* WcombT_h = (_Float16*)take((size_t)C_H * C_R * C_K, 2);  // BT for q_r
  _Float16* kvc_h    = (_Float16*)take((size_t)C_B * C_L * C_DC, 2);
  _Float16* kvcT_h   = (_Float16*)take((size_t)C_B * C_DC * C_L, 2); // BT for attn x V
  _Float16* WkrT_h   = (_Float16*)take((size_t)C_H * C_R * C_DC, 2); // BT for k_r
  float*    qr_f     = (float*)take((size_t)C_B * C_H * C_S * C_R, 4);
  float*    kr_f     = (float*)take((size_t)C_B * C_H * C_L * C_R, 4);
  _Float16* qr_h     = (_Float16*)take((size_t)C_B * C_H * C_S * C_R, 2);
  _Float16* kr_h     = (_Float16*)take((size_t)C_B * C_H * C_L * C_R, 2);
  _Float16* ctx_h    = (_Float16*)take((size_t)C_B * C_H * C_S * C_DC, 2);
  _Float16* wkckv_h  = (_Float16*)take((size_t)C_H * C_K * C_DC, 2); // BT for heads
  _Float16* heads_h  = (_Float16*)take((size_t)C_B * C_S * C_HK, 2);
  _Float16* Wout_h   = (_Float16*)take((size_t)C_DQ * C_HK, 2);      // BT for out-proj

  auto cvt = [&](const float* in, _Float16* o, long n) {
    cvt_kernel<<<dim3((unsigned)((n + 255) / 256)), dim3(256), 0, stream>>>(in, o, n);
  };
  auto cvtT = [&](const float* in, _Float16* o, int rows, int cols, int batch) {
    long n = (long)batch * rows * cols;
    cvt_t_kernel<<<dim3((unsigned)((n + 255) / 256)), dim3(256), 0, stream>>>(in, o, rows, cols, n);
  };
  auto gemm = [&](const _Float16* A, const _Float16* BT, void* Cp, bool outHalf,
                  int M, int N, int Kd, int lda, int ldbt, int ldc, long sA0,
                  long sA1, long sB0, long sB1, long sC0, long sC1, int nb0,
                  int nb1) {
    int tiles = (M / 16) * (N / 16);
    dim3 grid((unsigned)((tiles + 3) / 4), (unsigned)(nb0 * nb1));
    if (outHalf)
      wmma_gemm_kernel<_Float16><<<grid, dim3(128), 0, stream>>>(
          A, BT, (_Float16*)Cp, M, N, Kd, lda, ldbt, ldc, sA0, sA1, sB0, sB1, sC0, sC1, nb1);
    else
      wmma_gemm_kernel<float><<<grid, dim3(128), 0, stream>>>(
          A, BT, (float*)Cp, M, N, Kd, lda, ldbt, ldc, sA0, sA1, sB0, sB1, sC0, sC1, nb1);
  };
  auto gemm_big = [&](const _Float16* A, const _Float16* BT, void* Cp,
                      bool outHalf, int M, int N, int Kd, int lda, int ldbt,
                      int ldc) {
    dim3 grid((unsigned)(M / 128), (unsigned)(N / 64));
    if (outHalf)
      wmma_gemm_tiled_kernel<_Float16><<<grid, dim3(256), 0, stream>>>(
          A, BT, (_Float16*)Cp, M, N, Kd, lda, ldbt, ldc);
    else
      wmma_gemm_tiled_kernel<float><<<grid, dim3(256), 0, stream>>>(
          A, BT, (float*)Cp, M, N, Kd, lda, ldbt, ldc);
  };

  // --- operand conversions (all GEMM B operands stored transposed: N x K) ---
  cvt(hidden_q, hq_h, (long)C_B * C_S * C_DQ);
  cvt(kv_c, kvc_h, (long)C_B * C_L * C_DC);
  cvt(W_q, Wq_h, (long)C_HK * C_DQ);             // (e,d): already N x K
  cvt(W_out, Wout_h, (long)C_DQ * C_HK);         // (d,e): already N x K
  cvt(w_kc_q, wkcq_h, (long)C_H * C_K * C_DCQ);  // (k,q): N x K for WcombT GEMM
  cvt(w_kc_kv, wkckv_h, (long)C_H * C_K * C_DC); // (k,c): N x K for heads GEMM
  cvtT(W_qr, WqrT_h, C_DCQ, C_R, C_H);           // per-head (q,r) -> (r,q)
  cvtT(W_kr, WkrT_h, C_DC, C_R, C_H);            // per-head (c,r) -> (r,c)
  cvtT(kv_c, kvcT_h, C_L, C_DC, C_B);            // per-batch (l,c) -> (c,l)

  // --- q = hidden_q @ W_q^T : async-LDS tiled kernel, f16 out
  gemm_big(hq_h, Wq_h, q_h, true, C_B * C_S, C_HK, C_DQ, C_DQ, C_DQ, C_HK);

  // --- WcombT[h] (R x K) = W_qr[h]^T (R x DCQ) @ w_kc_q[h]^T
  gemm(WqrT_h, wkcq_h, WcombT_h, true, C_R, C_K, C_DCQ, C_DCQ, C_DCQ, C_K,
       0, (long)C_R * C_DCQ, 0, (long)C_K * C_DCQ, 0, (long)C_R * C_K, 1, C_H);

  // --- q_r[b,h] = q[b,:,h,:] (S x 64) @ Wcomb[h], BT = WcombT, f32 out
  gemm(q_h, WcombT_h, qr_f, false, C_S, C_R, C_K, C_HK, C_K, C_R,
       (long)C_S * C_HK, C_K, 0, (long)C_R * C_K,
       (long)C_H * C_S * C_R, (long)C_S * C_R, C_B, C_H);

  // --- k_r[b,h] = kv_c[b] (L x 128) @ W_kr[h], BT = W_kr^T, f32 out
  gemm(kvc_h, WkrT_h, kr_f, false, C_L, C_R, C_DC, C_DC, C_DC, C_R,
       (long)C_L * C_DC, 0, 0, (long)C_R * C_DC,
       (long)C_H * C_L * C_R, (long)C_L * C_R, C_B, C_H);

  // --- RoPE -> f16
  {
    long tq = (long)C_B * C_H * C_S * 32;
    rope_kernel<<<dim3((unsigned)((tq + 255) / 256)), dim3(256), 0, stream>>>(qr_f, qr_h, C_S, tq);
    long tk = (long)C_B * C_H * C_L * 32;
    rope_kernel<<<dim3((unsigned)((tk + 255) / 256)), dim3(256), 0, stream>>>(kr_f, kr_h, C_L, tk);
  }

  // --- fused attention: scores + softmax + attn x kv_c -> ctx f16
  {
    int blocks = C_B * C_H * (C_S / 128);    // 512 blocks, 8 waves each
    flash_attn_kernel<<<dim3((unsigned)blocks), dim3(256), 0, stream>>>(
        qr_h, kr_h, kvcT_h, ctx_h);
  }

  // --- heads[b,:,h,:] = ctx[b,h] (S x 128) @ w_kc_kv[h]^T, BT = w_kc_kv[h]
  gemm(ctx_h, wkckv_h, heads_h, true, C_S, C_K, C_DC, C_DC, C_DC, C_HK,
       (long)C_H * C_S * C_DC, (long)C_S * C_DC, 0, (long)C_K * C_DC,
       (long)C_S * C_HK, C_K, C_B, C_H);

  // --- out = heads @ W_out^T : async-LDS tiled kernel, f32 out
  gemm_big(heads_h, Wout_h, out, false, C_B * C_S, C_DQ, C_HK, C_HK, C_HK, C_DQ);
}
